// FinancialForecastModel_62732292326028
// MI455X (gfx1250) — matrix operand order repaired and checked by reading, not compiled, not run on hardware
//
#include <hip/hip_runtime.h>
#include <hip/hip_bf16.h>
#include <cstdint>

// MI455X (gfx1250) implementation of the coRNN financial-forecast model.
// All width-3 convs are 3 shifted GEMMs on v_wmma_f32_16x16x32_f16 (f16 in,
// f32 accumulate). Activations in [b][l][c] f16 => WMMA B operand is one
// contiguous 32B v16h load per lane; weights pre-swizzled to a k=16h+s A-operand
// lane layout => A is one contiguous 32B load per lane. The 50-step recurrence
// is one fused kernel per step; each wave owns two adjacent l-tiles so each
// weight A-tile is loaded once and reused for 2 N-tiles x 2 convs (48 WMMAs
// per wave, 2048 waves/step). f32 master state ping-pongs in L2.

typedef __attribute__((ext_vector_type(16))) _Float16 v16h;
typedef __attribute__((ext_vector_type(8)))  float    v8f;

#define B_N   16
#define L_N   512
#define CIN_  8
#define CMID_ 64
#define COUT_ 128
#define T_N   50
#define HZN_  8
#define DT_C  0.5f

// ---------------- weight swizzle: torch (Cout,Cin,3) f32 -> WMMA A-operand tiles ----
// tile order: [k][chunk][coTile][lane][16 halves]; 16-bit A 16x32 in contiguous-k order (B matches):
// lane m (M row), hi=lane>>4 selects K half: halves j -> K=hi*16+j,
// all offset by chunk*32.
__global__ void prep_weights(const float* __restrict__ w, _Float16* __restrict__ out,
                             int Cout, int CinReal, int CinPad) {
  int gid = blockIdx.x * blockDim.x + threadIdx.x;
  int nCoT = Cout >> 4, nChunk = CinPad >> 5;
  int total = 3 * nChunk * nCoT * 32;
  if (gid >= total) return;
  int lane = gid & 31, tile = gid >> 5;
  int coT = tile % nCoT;
  int chunk = (tile / nCoT) % nChunk;
  int k = tile / (nCoT * nChunk);
  int m = lane & 15, hi = lane >> 4;
  int co = coT * 16 + m;
  _Float16* dst = out + ((size_t)tile * 32 + lane) * 16;
  for (int j = 0; j < 16; ++j) {
    int K = chunk * 32 + hi * 16 + j;
    float v = (K < CinReal) ? w[((size_t)co * CinReal + K) * 3 + k] : 0.0f;
    dst[j] = (_Float16)v;
  }
}

// x (B,CIN,L) f32 -> x16 [b][l][32] f16, channels zero-padded 8->32
__global__ void pack_x(const float* __restrict__ x, _Float16* __restrict__ out) {
  int gid = blockIdx.x * blockDim.x + threadIdx.x;
  if (gid >= B_N * L_N * 32) return;
  int c = gid & 31;
  int l = (gid >> 5) % L_N;
  int b = gid / (32 * L_N);
  float v = (c < CIN_) ? x[((size_t)b * CIN_ + c) * L_N + l] : 0.0f;
  out[gid] = (_Float16)v;
}

// generic conv1d(width 3, same pad) + bias + activation via WMMA (encoder path).
// actMode: 0=none, 1=relu, 2=tanh. out16 ([b][l][Cout] f16) / out32 nullable.
__global__ void conv_wmma(const _Float16* __restrict__ actIn, int CinPad,
                          const _Float16* __restrict__ wt, const float* __restrict__ bias,
                          int Cout, int actMode,
                          _Float16* __restrict__ out16, float* __restrict__ out32) {
  int tid = blockIdx.x * blockDim.x + threadIdx.x;
  int lane = tid & 31, wave = tid >> 5;
  int nCoT = Cout >> 4, nLT = L_N >> 4, nChunk = CinPad >> 5;
  int nTiles = B_N * nCoT * nLT;
  if (wave >= nTiles) return;              // wave-uniform; EXEC all-1s at WMMA
  int lT = wave % nLT;
  int coT = (wave / nLT) % nCoT;
  int b = wave / (nLT * nCoT);
  int n = lane & 15, hi = lane >> 4;
  v8f acc = {};
  for (int k = 0; k < 3; ++k) {
    int l = lT * 16 + n + k - 1;
    bool ok = (l >= 0) && (l < L_N);
    const _Float16* bbase = actIn + (((size_t)b * L_N + (ok ? l : 0)) * CinPad + hi * 16);
    for (int ch = 0; ch < nChunk; ++ch) {
      v16h a = *(const v16h*)(wt + (((size_t)(k * nChunk + ch) * nCoT + coT) * 32 + lane) * 16);
      v16h bm = {};
      if (ok) bm = *(const v16h*)(bbase + ch * 32);
      acc = __builtin_amdgcn_wmma_f32_16x16x32_f16(false, a, false, bm,
                                                   (short)0, acc, false, false);
    }
  }
  int lOut = lT * 16 + n;
  for (int r = 0; r < 8; ++r) {
    int co = coT * 16 + r + hi * 8;
    float v = acc[r] + bias[co];
    if (actMode == 1) v = fmaxf(v, 0.0f);
    else if (actMode == 2) v = tanhf(v);
    size_t idx = ((size_t)b * L_N + lOut) * Cout + co;
    if (out32) out32[idx] = v;
    if (out16) out16[idx] = (_Float16)v;
  }
}

// one coRNN step: pre = tanh(conv(hy,cwy)+conv(hz,cwz)+xdrive);
// hz' = hz + DT*(pre - omega*hy - alpha*hz); hy' = hy + DT*hz'; y_seq[t] = hy'
// Each wave computes a 16(co) x 32(l) output strip (two 16x16 N-tiles) so every
// weight A-tile load is reused 2x (and across both convs): 24 A-loads,
// 48 B-loads, 48 WMMAs per wave; 2048 waves per step.
__global__ void corrn_step(const _Float16* __restrict__ hy16, const _Float16* __restrict__ hz16,
                           const float* __restrict__ hyF, const float* __restrict__ hzF,
                           const _Float16* __restrict__ wty, const _Float16* __restrict__ wtz,
                           const float* __restrict__ cby, const float* __restrict__ cbz,
                           const float* __restrict__ omega, const float* __restrict__ alpha,
                           const float* __restrict__ xdrive,
                           float* __restrict__ hyN, float* __restrict__ hzN,
                           _Float16* __restrict__ hy16N, _Float16* __restrict__ hz16N,
                           float* __restrict__ yseq, int t) {
  int tid = blockIdx.x * blockDim.x + threadIdx.x;
  int lane = tid & 31, wave = tid >> 5;
  const int nCoT = COUT_ >> 4;          // 8
  const int nLP  = L_N >> 5;            // 16 pairs of l-tiles
  int nStrips = B_N * nCoT * nLP;       // 2048 waves
  if (wave >= nStrips) return;
  int lP  = wave % nLP;                 // consecutive waves share (b,coT) -> WGP$ hits on A
  int coT = (wave / nLP) % nCoT;
  int b   = wave / (nLP * nCoT);
  int n = lane & 15, hi = lane >> 4;
  v8f acc0 = {}, acc1 = {};
#pragma unroll
  for (int k = 0; k < 3; ++k) {
#pragma unroll
    for (int ch = 0; ch < (COUT_ >> 5); ++ch) {
      size_t wofs = (((size_t)(k * (COUT_ >> 5) + ch) * nCoT + coT) * 32 + lane) * 16;
      v16h ay = *(const v16h*)(wty + wofs);
      v16h az = *(const v16h*)(wtz + wofs);
#pragma unroll
      for (int s = 0; s < 2; ++s) {
        int l = lP * 32 + s * 16 + n + k - 1;
        bool ok = (l >= 0) && (l < L_N);
        size_t bofs = ((size_t)b * L_N + (ok ? l : 0)) * COUT_ + hi * 16 + ch * 32;
        v16h by = {}, bz = {};
        if (ok) {
          by = *(const v16h*)(hy16 + bofs);
          bz = *(const v16h*)(hz16 + bofs);
        }
        if (s == 0) {
          acc0 = __builtin_amdgcn_wmma_f32_16x16x32_f16(false, ay, false, by,
                                                        (short)0, acc0, false, false);
          acc0 = __builtin_amdgcn_wmma_f32_16x16x32_f16(false, az, false, bz,
                                                        (short)0, acc0, false, false);
        } else {
          acc1 = __builtin_amdgcn_wmma_f32_16x16x32_f16(false, ay, false, by,
                                                        (short)0, acc1, false, false);
          acc1 = __builtin_amdgcn_wmma_f32_16x16x32_f16(false, az, false, bz,
                                                        (short)0, acc1, false, false);
        }
      }
    }
  }
#pragma unroll
  for (int s = 0; s < 2; ++s) {
    int lOut = lP * 32 + s * 16 + n;
    const v8f& acc = (s == 0) ? acc0 : acc1;
#pragma unroll
    for (int r = 0; r < 8; ++r) {
      int co = coT * 16 + r + hi * 8;
      size_t idx = ((size_t)b * L_N + lOut) * COUT_ + co;
      float hy = hyF[idx], hz = hzF[idx];
      float pre = tanhf(acc[r] + cby[co] + cbz[co] + xdrive[idx]);
      float hzn = hz + DT_C * (pre - omega[idx] * hy - alpha[idx] * hz);
      float hyn = hy + DT_C * hzn;
      hyN[idx] = hyn;
      hzN[idx] = hzn;
      hy16N[idx] = (_Float16)hyn;
      hz16N[idx] = (_Float16)hzn;
      yseq[(((size_t)b * T_N + t) * COUT_ + co) * L_N + lOut] = hyn;
    }
  }
}

__global__ void zero_state(float* __restrict__ hz, _Float16* __restrict__ hz16) {
  int gid = blockIdx.x * blockDim.x + threadIdx.x;
  if (gid >= B_N * L_N * COUT_) return;
  hz[gid] = 0.0f;
  hz16[gid] = (_Float16)0.0f;
}

// feat[b][co] = mean_l hy[b][l][co]  (row reads are fully coalesced)
__global__ void reduce_feat(const float* __restrict__ hy, float* __restrict__ feat) {
  int b = blockIdx.x, co = threadIdx.x; // 128 threads
  float s = 0.0f;
  for (int l = 0; l < L_N; ++l) s += hy[((size_t)b * L_N + l) * COUT_ + co];
  feat[b * COUT_ + co] = s * (1.0f / L_N);
}

__global__ void readout_mlp(const float* __restrict__ feat,
                            const float* __restrict__ fw1, const float* __restrict__ fb1,
                            const float* __restrict__ fw2, const float* __restrict__ fb2,
                            const float* __restrict__ fw3, const float* __restrict__ fb3,
                            float* __restrict__ pred) {
  __shared__ float f[COUT_], h1[64], h2[32];
  int b = blockIdx.x, t = threadIdx.x; // 128 threads
  f[t] = feat[b * COUT_ + t];
  __syncthreads();
  if (t < 64) {
    float s = fb1[t];
    for (int i = 0; i < COUT_; ++i) s += f[i] * fw1[t * COUT_ + i];
    h1[t] = fmaxf(s, 0.0f);
  }
  __syncthreads();
  if (t < 32) {
    float s = fb2[t];
    for (int i = 0; i < 64; ++i) s += h1[i] * fw2[t * 64 + i];
    h2[t] = fmaxf(s, 0.0f);
  }
  __syncthreads();
  if (t < HZN_) {
    float s = fb3[t];
    for (int i = 0; i < 32; ++i) s += h2[i] * fw3[t * 32 + i];
    pred[b * HZN_ + t] = s;
  }
}

static inline size_t alignup(size_t x) { return (x + 255) & ~(size_t)255; }

extern "C" void kernel_launch(void* const* d_in, const int* in_sizes, int n_in,
                              void* d_out, int out_size, void* d_ws, size_t ws_size,
                              hipStream_t stream) {
  (void)in_sizes; (void)n_in; (void)out_size; (void)ws_size;
  const float* x   = (const float*)d_in[0];
  const float* ow1 = (const float*)d_in[1];  const float* ob1 = (const float*)d_in[2];
  const float* ow2 = (const float*)d_in[3];  const float* ob2 = (const float*)d_in[4];
  const float* ow3 = (const float*)d_in[5];  const float* ob3 = (const float*)d_in[6];
  const float* aw1 = (const float*)d_in[7];  const float* ab1 = (const float*)d_in[8];
  const float* aw2 = (const float*)d_in[9];  const float* ab2 = (const float*)d_in[10];
  const float* aw3 = (const float*)d_in[11]; const float* ab3 = (const float*)d_in[12];
  const float* hw1 = (const float*)d_in[13]; const float* hb1 = (const float*)d_in[14];
  const float* hw2 = (const float*)d_in[15]; const float* hb2 = (const float*)d_in[16];
  const float* hw3 = (const float*)d_in[17]; const float* hb3 = (const float*)d_in[18];
  const float* hw4 = (const float*)d_in[19]; const float* hb4 = (const float*)d_in[20];
  const float* cwy = (const float*)d_in[21]; const float* cby = (const float*)d_in[22];
  const float* cwz = (const float*)d_in[23]; const float* cbz = (const float*)d_in[24];
  const float* cwx = (const float*)d_in[25]; const float* cbx = (const float*)d_in[26];
  const float* fw1 = (const float*)d_in[27]; const float* fb1 = (const float*)d_in[28];
  const float* fw2 = (const float*)d_in[29]; const float* fb2 = (const float*)d_in[30];
  const float* fw3 = (const float*)d_in[31]; const float* fb3 = (const float*)d_in[32];

  float* pred = (float*)d_out;
  float* yseq = (float*)d_out + B_N * HZN_;

  // -------- workspace layout --------
  char* ws = (char*)d_ws;
  size_t off = 0;
  auto alloc = [&](size_t bytes) { void* p = ws + off; off = alignup(off + bytes); return p; };

  _Float16* x16 = (_Float16*)alloc((size_t)B_N * L_N * 32 * 2);
  _Float16* t1  = (_Float16*)alloc((size_t)B_N * L_N * CMID_ * 2);
  _Float16* t2  = (_Float16*)alloc((size_t)B_N * L_N * CMID_ * 2);
  float* omega32  = (float*)alloc((size_t)B_N * L_N * COUT_ * 4);
  float* alpha32  = (float*)alloc((size_t)B_N * L_N * COUT_ * 4);
  float* xdrive32 = (float*)alloc((size_t)B_N * L_N * COUT_ * 4);
  float* hyF[2]   = { (float*)alloc((size_t)B_N * L_N * COUT_ * 4),
                      (float*)alloc((size_t)B_N * L_N * COUT_ * 4) };
  float* hzF[2]   = { (float*)alloc((size_t)B_N * L_N * COUT_ * 4),
                      (float*)alloc((size_t)B_N * L_N * COUT_ * 4) };
  _Float16* hy16[2] = { (_Float16*)alloc((size_t)B_N * L_N * COUT_ * 2),
                        (_Float16*)alloc((size_t)B_N * L_N * COUT_ * 2) };
  _Float16* hz16[2] = { (_Float16*)alloc((size_t)B_N * L_N * COUT_ * 2),
                        (_Float16*)alloc((size_t)B_N * L_N * COUT_ * 2) };
  float* feat = (float*)alloc((size_t)B_N * COUT_ * 4);

  auto wbytes = [](int CinPad, int Cout) { return (size_t)3 * (CinPad / 32) * (Cout / 16) * 32 * 16 * 2; };
  _Float16* wt_ow1 = (_Float16*)alloc(wbytes(32, 64));
  _Float16* wt_ow2 = (_Float16*)alloc(wbytes(64, 64));
  _Float16* wt_ow3 = (_Float16*)alloc(wbytes(64, 128));
  _Float16* wt_aw1 = (_Float16*)alloc(wbytes(32, 64));
  _Float16* wt_aw2 = (_Float16*)alloc(wbytes(64, 64));
  _Float16* wt_aw3 = (_Float16*)alloc(wbytes(64, 128));
  _Float16* wt_hw1 = (_Float16*)alloc(wbytes(32, 64));
  _Float16* wt_hw2 = (_Float16*)alloc(wbytes(64, 64));
  _Float16* wt_hw3 = (_Float16*)alloc(wbytes(64, 64));
  _Float16* wt_hw4 = (_Float16*)alloc(wbytes(64, 128));
  _Float16* wt_cwx = (_Float16*)alloc(wbytes(32, 128));
  _Float16* wt_cwy = (_Float16*)alloc(wbytes(128, 128));
  _Float16* wt_cwz = (_Float16*)alloc(wbytes(128, 128));

  // -------- weight prep + input pack --------
  auto prep = [&](const float* w, _Float16* outw, int Cout, int CinReal, int CinPad) {
    int total = 3 * (CinPad / 32) * (Cout / 16) * 32;
    prep_weights<<<(total + 127) / 128, 128, 0, stream>>>(w, outw, Cout, CinReal, CinPad);
  };
  prep(ow1, wt_ow1, 64, CIN_, 32);   prep(ow2, wt_ow2, 64, 64, 64);   prep(ow3, wt_ow3, 128, 64, 64);
  prep(aw1, wt_aw1, 64, CIN_, 32);   prep(aw2, wt_aw2, 64, 64, 64);   prep(aw3, wt_aw3, 128, 64, 64);
  prep(hw1, wt_hw1, 64, CIN_, 32);   prep(hw2, wt_hw2, 64, 64, 64);   prep(hw3, wt_hw3, 64, 64, 64);
  prep(hw4, wt_hw4, 128, 64, 64);    prep(cwx, wt_cwx, 128, CIN_, 32);
  prep(cwy, wt_cwy, 128, 128, 128);  prep(cwz, wt_cwz, 128, 128, 128);

  pack_x<<<(B_N * L_N * 32 + 127) / 128, 128, 0, stream>>>(x, x16);

  auto conv = [&](const _Float16* in, int CinPad, const _Float16* wt, const float* bias,
                  int Cout, int act, _Float16* o16, float* o32) {
    int waves = B_N * (Cout / 16) * (L_N / 16); // multiple of 4
    conv_wmma<<<(waves * 32) / 128, 128, 0, stream>>>(in, CinPad, wt, bias, Cout, act, o16, o32);
  };

  // encoders
  conv(x16, 32, wt_ow1, ob1, 64, 1, t1, nullptr);
  conv(t1, 64, wt_ow2, ob2, 64, 1, t2, nullptr);
  conv(t2, 64, wt_ow3, ob3, 128, 1, nullptr, omega32);

  conv(x16, 32, wt_aw1, ab1, 64, 1, t1, nullptr);
  conv(t1, 64, wt_aw2, ab2, 64, 1, t2, nullptr);
  conv(t2, 64, wt_aw3, ab3, 128, 1, nullptr, alpha32);

  conv(x16, 32, wt_hw1, hb1, 64, 1, t1, nullptr);
  conv(t1, 64, wt_hw2, hb2, 64, 1, t2, nullptr);
  conv(t2, 64, wt_hw3, hb3, 64, 1, t1, nullptr);
  conv(t1, 64, wt_hw4, hb4, 128, 2, hy16[0], hyF[0]);   // hy0 = tanh(...)

  conv(x16, 32, wt_cwx, cbx, 128, 0, nullptr, xdrive32); // x_drive (constant over steps)

  zero_state<<<(B_N * L_N * COUT_ + 127) / 128, 128, 0, stream>>>(hzF[0], hz16[0]);

  // -------- recurrence --------
  int stepWaves = B_N * (COUT_ / 16) * (L_N / 32); // 2048 strip-waves
  for (int t = 0; t < T_N; ++t) {
    int cur = t & 1, nxt = (t + 1) & 1;
    corrn_step<<<(stepWaves * 32) / 128, 128, 0, stream>>>(
        hy16[cur], hz16[cur], hyF[cur], hzF[cur],
        wt_cwy, wt_cwz, cby, cbz,
        omega32, alpha32, xdrive32,
        hyF[nxt], hzF[nxt], hy16[nxt], hz16[nxt],
        yseq, t);
  }

  // -------- readout --------
  reduce_feat<<<B_N, COUT_, 0, stream>>>(hyF[T_N & 1], feat);
  readout_mlp<<<B_N, COUT_, 0, stream>>>(feat, fw1, fb1, fw2, fb2, fw3, fb3, pred);
}
